// RCVResblock_81544249081945
// MI455X (gfx1250) — matrix operand order repaired and checked by reading, not compiled, not run on hardware
//
#include <hip/hip_runtime.h>

typedef __bf16 bf16;
typedef __attribute__((ext_vector_type(16))) __bf16 v16bf;
typedef __attribute__((ext_vector_type(8)))  float  v8f;

constexpr int FDIM = 512;   // feature dim
constexpr int MT   = 32;    // rows per workgroup (2 m-tiles of 16)
constexpr int NW   = 8;     // waves per workgroup
constexpr float LN_EPS = 1e-5f;

// ---------------------------------------------------------------------------
// Weight prep: fp32 -> (hi, lo) bf16 split, with optional sign flip so the
// imaginary-part weights are pre-negated (GEMM becomes pure accumulation).
// ---------------------------------------------------------------------------
__global__ void prep_weights(const float* __restrict__ w, float sign,
                             bf16* __restrict__ hi, bf16* __restrict__ lo) {
    int i = blockIdx.x * blockDim.x + threadIdx.x;
    if (i < FDIM * FDIM) {
        float f = sign * w[i];
        bf16  h = (bf16)f;
        hi[i] = h;
        lo[i] = (bf16)(f - (float)h);
    }
}

// Convert 8 consecutive floats into positions [base, base+8) of hi/lo frags.
__device__ __forceinline__ void cvt8(float4 a, float4 b, v16bf& hi, v16bf& lo, int base) {
    float f[8] = {a.x, a.y, a.z, a.w, b.x, b.y, b.z, b.w};
#pragma unroll
    for (int i = 0; i < 8; ++i) {
        bf16 h = (bf16)f[i];
        hi[base + i] = h;
        lo[base + i] = (bf16)(f[i] - (float)h);
    }
}

// A-fragment (16x32 bf16, contiguous k order used here, same as B):
// lane half 0: K = k0+0..15 (contiguous); half 1: K = k0+16..31.
__device__ __forceinline__ void load_afrag(const float* __restrict__ rowp, int k0, int lhalf,
                                           v16bf& hi, v16bf& lo) {
    const float4* p0 = (const float4*)(rowp + k0 + 16 * lhalf);
    const float4* p1 = (const float4*)(rowp + k0 + 16 * lhalf + 8);
    cvt8(p0[0], p0[1], hi, lo, 0);
    cvt8(p1[0], p1[1], hi, lo, 8);
}

// Row-stat reduction: per-wave shuffle reduce over 16 columns, cross-wave
// combine through LDS. s_red: [MT][NW][2], s_stat: [MT][2] (mu, rstd).
__device__ __forceinline__ void row_stats(v8f acc[2][4], int wave, int l15, int lhalf,
                                          float* s_red, float* s_stat) {
#pragma unroll
    for (int mt = 0; mt < 2; ++mt) {
#pragma unroll
        for (int r = 0; r < 8; ++r) {
            float p = 0.f, q = 0.f;
#pragma unroll
            for (int nt = 0; nt < 4; ++nt) {
                float v = acc[mt][nt][r];
                p += v;
                q += v * v;
            }
#pragma unroll
            for (int off = 1; off < 16; off <<= 1) {
                p += __shfl_xor(p, off, 32);
                q += __shfl_xor(q, off, 32);
            }
            if (l15 == 0) {
                int row = mt * 16 + 8 * lhalf + r;
                s_red[(row * NW + wave) * 2 + 0] = p;
                s_red[(row * NW + wave) * 2 + 1] = q;
            }
        }
    }
    __syncthreads();
    if (threadIdx.x < MT) {
        float S = 0.f, Q = 0.f;
#pragma unroll
        for (int w = 0; w < NW; ++w) {
            S += s_red[(threadIdx.x * NW + w) * 2 + 0];
            Q += s_red[(threadIdx.x * NW + w) * 2 + 1];
        }
        float mu  = S * (1.0f / FDIM);
        float var = Q * (1.0f / FDIM) - mu * mu;
        s_stat[threadIdx.x * 2 + 0] = mu;
        s_stat[threadIdx.x * 2 + 1] = rsqrtf(var + LN_EPS);
    }
    __syncthreads();
}

// ---------------------------------------------------------------------------
// Fused: GEMM1(K=1024, bf16x3) -> LN -> soft-threshold -> GEMM2 -> LN ->
// residual -> soft-threshold -> store. One WG = 32 rows x 512 cols.
// ---------------------------------------------------------------------------
__global__ __launch_bounds__(256) void rcv_fused(
    const float* __restrict__ x,
    const bf16* __restrict__ w1r_hi, const bf16* __restrict__ w1r_lo,
    const bf16* __restrict__ w1i_hi, const bf16* __restrict__ w1i_lo,   // pre-negated
    const bf16* __restrict__ w2r_hi, const bf16* __restrict__ w2r_lo,
    const bf16* __restrict__ w2i_hi, const bf16* __restrict__ w2i_lo,   // pre-negated
    const float* __restrict__ g1r, const float* __restrict__ b1r,
    const float* __restrict__ g1i, const float* __restrict__ b1i,
    const float* __restrict__ theta1,
    const float* __restrict__ g2r, const float* __restrict__ b2r,
    const float* __restrict__ g2i, const float* __restrict__ b2i,
    const float* __restrict__ theta_f,
    float* __restrict__ out) {

    const int tid   = threadIdx.x;
    const int wave  = tid >> 5;
    const int lane  = tid & 31;
    const int l15   = lane & 15;
    const int lhalf = lane >> 4;
    const int row0  = blockIdx.x * MT;
    const int ncol0 = wave * 64;

    extern __shared__ __align__(16) char smem_raw[];
    float* s_a    = (float*)smem_raw;          // [MT][FDIM]
    float* s_b    = s_a + MT * FDIM;           // [MT][FDIM]
    float* s_red  = s_b + MT * FDIM;           // [MT][NW][2]
    float* s_stat = s_red + MT * NW * 2;       // [MT][2]

    const float th1 = theta1[0];
    const float thf = theta_f[0];

    v8f acc[2][4];
#pragma unroll
    for (int mt = 0; mt < 2; ++mt)
#pragma unroll
        for (int nt = 0; nt < 4; ++nt)
            acc[mt][nt] = (v8f){0.f, 0.f, 0.f, 0.f, 0.f, 0.f, 0.f, 0.f};

    // ------------------- GEMM1: y = re@W1r.T + im@(-W1i).T -------------------
#pragma unroll 1
    for (int kc = 0; kc < 32; ++kc) {
        const int   ch = kc >> 4;            // 0: re/W1r, 1: im/-W1i
        const int   kl = (kc & 15) * 32;     // K offset within the 512 half
        const bf16* wh = ch ? w1i_hi : w1r_hi;
        const bf16* wl = ch ? w1i_lo : w1r_lo;

        v16bf ahi[2], alo[2];
#pragma unroll
        for (int mt = 0; mt < 2; ++mt) {
            size_t grow = (size_t)(row0 + mt * 16 + l15);
            load_afrag(x + (grow * 2 + ch) * FDIM, kl, lhalf, ahi[mt], alo[mt]);
        }
#pragma unroll
        for (int nt = 0; nt < 4; ++nt) {
            size_t n    = (size_t)(ncol0 + nt * 16 + l15);
            int    koff = kl + 16 * lhalf;
            v16bf bhi = *(const v16bf*)(wh + n * FDIM + koff);
            v16bf blo = *(const v16bf*)(wl + n * FDIM + koff);
#pragma unroll
            for (int mt = 0; mt < 2; ++mt) {
                acc[mt][nt] = __builtin_amdgcn_wmma_f32_16x16x32_bf16(false, ahi[mt], false, bhi, (short)0, acc[mt][nt], false, false);
                acc[mt][nt] = __builtin_amdgcn_wmma_f32_16x16x32_bf16(false, alo[mt], false, bhi, (short)0, acc[mt][nt], false, false);
                acc[mt][nt] = __builtin_amdgcn_wmma_f32_16x16x32_bf16(false, ahi[mt], false, blo, (short)0, acc[mt][nt], false, false);
            }
        }
    }

    // ------------------- LN1 + soft-threshold -> LDS -------------------
    row_stats(acc, wave, l15, lhalf, s_red, s_stat);

#pragma unroll
    for (int nt = 0; nt < 4; ++nt) {
        int   col = ncol0 + nt * 16 + l15;
        float gr = g1r[col], br = b1r[col], gi = g1i[col], bi = b1i[col];
#pragma unroll
        for (int mt = 0; mt < 2; ++mt)
#pragma unroll
            for (int r = 0; r < 8; ++r) {
                int   row = mt * 16 + 8 * lhalf + r;
                float mu  = s_stat[row * 2 + 0];
                float rs  = s_stat[row * 2 + 1];
                float yn  = (acc[mt][nt][r] - mu) * rs;
                float a   = yn * gr + br;
                float b   = yn * gi + bi;
                float ab  = sqrtf(a * a + b * b);
                float Bv  = fmaxf(ab - th1, 0.f);
                float sc  = ab > 0.f ? Bv / ab : 0.f;
                s_a[row * FDIM + col] = a * sc;
                s_b[row * FDIM + col] = b * sc;
            }
    }
    __syncthreads();

    // ------------------- GEMM2: z = a@W2r.T + b@(-W2i).T -------------------
#pragma unroll
    for (int mt = 0; mt < 2; ++mt)
#pragma unroll
        for (int nt = 0; nt < 4; ++nt)
            acc[mt][nt] = (v8f){0.f, 0.f, 0.f, 0.f, 0.f, 0.f, 0.f, 0.f};

#pragma unroll 1
    for (int kc = 0; kc < 32; ++kc) {
        const int    ch  = kc >> 4;
        const int    kl  = (kc & 15) * 32;
        const bf16*  wh  = ch ? w2i_hi : w2r_hi;
        const bf16*  wl  = ch ? w2i_lo : w2r_lo;
        const float* src = ch ? s_b : s_a;

        v16bf ahi[2], alo[2];
#pragma unroll
        for (int mt = 0; mt < 2; ++mt)
            load_afrag(src + (mt * 16 + l15) * FDIM, kl, lhalf, ahi[mt], alo[mt]);

#pragma unroll
        for (int nt = 0; nt < 4; ++nt) {
            size_t n    = (size_t)(ncol0 + nt * 16 + l15);
            int    koff = kl + 16 * lhalf;
            v16bf bhi = *(const v16bf*)(wh + n * FDIM + koff);
            v16bf blo = *(const v16bf*)(wl + n * FDIM + koff);
#pragma unroll
            for (int mt = 0; mt < 2; ++mt) {
                acc[mt][nt] = __builtin_amdgcn_wmma_f32_16x16x32_bf16(false, ahi[mt], false, bhi, (short)0, acc[mt][nt], false, false);
                acc[mt][nt] = __builtin_amdgcn_wmma_f32_16x16x32_bf16(false, alo[mt], false, bhi, (short)0, acc[mt][nt], false, false);
                acc[mt][nt] = __builtin_amdgcn_wmma_f32_16x16x32_bf16(false, ahi[mt], false, blo, (short)0, acc[mt][nt], false, false);
            }
        }
    }
    __syncthreads();   // LDS (s_a/s_b) no longer read; reuse s_red/s_stat next

    // ------------------- LN2 + residual + soft-threshold -> out -------------------
    row_stats(acc, wave, l15, lhalf, s_red, s_stat);

#pragma unroll
    for (int nt = 0; nt < 4; ++nt) {
        int   col = ncol0 + nt * 16 + l15;
        float gr = g2r[col], br = b2r[col], gi = g2i[col], bi = b2i[col];
#pragma unroll
        for (int mt = 0; mt < 2; ++mt)
#pragma unroll
            for (int r = 0; r < 8; ++r) {
                int    row  = mt * 16 + 8 * lhalf + r;
                size_t grow = (size_t)(row0 + row);
                float  mu   = s_stat[row * 2 + 0];
                float  rs   = s_stat[row * 2 + 1];
                float  zn   = (acc[mt][nt][r] - mu) * rs;
                float  a    = zn * gr + br + x[(grow * 2 + 0) * FDIM + col];
                float  b    = zn * gi + bi + x[(grow * 2 + 1) * FDIM + col];
                float  ab   = sqrtf(a * a + b * b);
                float  Bv   = fmaxf(ab - thf, 0.f);
                float  sc   = ab > 0.f ? Bv / ab : 0.f;
                out[(grow * 2 + 0) * FDIM + col] = a * sc;
                out[(grow * 2 + 1) * FDIM + col] = b * sc;
            }
    }
}

extern "C" void kernel_launch(void* const* d_in, const int* in_sizes, int n_in,
                              void* d_out, int out_size, void* d_ws, size_t ws_size,
                              hipStream_t stream) {
    const float* x      = (const float*)d_in[0];
    const float* W1r    = (const float*)d_in[1];
    const float* W1i    = (const float*)d_in[2];
    const float* g1r    = (const float*)d_in[3];
    const float* b1r    = (const float*)d_in[4];
    const float* g1i    = (const float*)d_in[5];
    const float* b1i    = (const float*)d_in[6];
    const float* theta1 = (const float*)d_in[7];
    const float* W2r    = (const float*)d_in[8];
    const float* W2i    = (const float*)d_in[9];
    const float* g2r    = (const float*)d_in[10];
    const float* b2r    = (const float*)d_in[11];
    const float* g2i    = (const float*)d_in[12];
    const float* b2i    = (const float*)d_in[13];
    const float* thetaf = (const float*)d_in[14];

    const int WSZ = FDIM * FDIM;
    bf16* ws = (bf16*)d_ws;
    bf16 *w1r_hi = ws + 0 * WSZ, *w1r_lo = ws + 1 * WSZ;
    bf16 *w1i_hi = ws + 2 * WSZ, *w1i_lo = ws + 3 * WSZ;
    bf16 *w2r_hi = ws + 4 * WSZ, *w2r_lo = ws + 5 * WSZ;
    bf16 *w2i_hi = ws + 6 * WSZ, *w2i_lo = ws + 7 * WSZ;

    dim3 pb(256), pg((WSZ + 255) / 256);
    prep_weights<<<pg, pb, 0, stream>>>(W1r,  1.f, w1r_hi, w1r_lo);
    prep_weights<<<pg, pb, 0, stream>>>(W1i, -1.f, w1i_hi, w1i_lo);
    prep_weights<<<pg, pb, 0, stream>>>(W2r,  1.f, w2r_hi, w2r_lo);
    prep_weights<<<pg, pb, 0, stream>>>(W2i, -1.f, w2i_hi, w2i_lo);

    const int Brows = in_sizes[0] / (2 * FDIM);   // 65536
    size_t shmem = (size_t)(2 * MT * FDIM + MT * NW * 2 + MT * 2) * sizeof(float);
    rcv_fused<<<Brows / MT, 256, shmem, stream>>>(
        x, w1r_hi, w1r_lo, w1i_hi, w1i_lo, w2r_hi, w2r_lo, w2i_hi, w2i_lo,
        g1r, b1r, g1i, b1i, theta1, g2r, b2r, g2i, b2i, thetaf,
        (float*)d_out);
}
